// JointVAE_1297080123931
// MI455X (gfx1250) — matrix operand order repaired and checked by reading, not compiled, not run on hardware
//
#include <hip/hip_runtime.h>

typedef __attribute__((ext_vector_type(16))) __bf16 v16bf;
typedef __attribute__((ext_vector_type(8)))  __bf16 v8bf;
typedef __attribute__((ext_vector_type(4)))  __bf16 v4bf;
typedef __attribute__((ext_vector_type(8)))  float  v8f;

#define TILE_M 128
#define TILE_N 128
#define TILE_K 32
#define LDSK   40   // padded bf16 K-stride: 80B rows -> conflict-free, 16B aligned

enum { EPI_RELU_BIAS = 0, EPI_RES = 1, EPI_BIAS = 2, EPI_ARGMIN = 3 };

// monotone float -> u32 key (ascending unsigned order == ascending float order)
__device__ __forceinline__ unsigned float_key(float f) {
  unsigned u = __float_as_uint(f);
  return (u & 0x80000000u) ? ~u : (u | 0x80000000u);
}

__device__ __forceinline__ unsigned long long shfl_xor_u64(unsigned long long v, int m) {
  unsigned lo = (unsigned)__shfl_xor((int)(unsigned)v, m, 32);
  unsigned hi = (unsigned)__shfl_xor((int)(unsigned)(v >> 32), m, 32);
  return ((unsigned long long)hi << 32) | lo;
}

// C = epilogue(A[f32 MxK] x B + bias), bf16 WMMA, f32 accumulate.
// B_NMAJOR=true : Bsrc is [N][K] row-major (codebook case, direct copy to LDS)
// B_NMAJOR=false: Bsrc is [K][N] row-major (weight matrices, transpose on LDS store)
template<int EPI, bool B_NMAJOR>
__global__ __launch_bounds__(256)
void gemm_bf16_wmma(const float* __restrict__ A,
                    const float* __restrict__ Bsrc,
                    const float* __restrict__ bias,
                    const float* __restrict__ Cadd,
                    float* __restrict__ Out,
                    const float* __restrict__ rowsq,
                    unsigned long long* __restrict__ amin,
                    int M, int N, int K)
{
  __shared__ __bf16 As[TILE_M][LDSK];
  __shared__ __bf16 Bs[TILE_N][LDSK];   // stored n-major: [n][k]

  const int tid   = threadIdx.x;
  const int lane  = tid & 31;
  const int wave  = tid >> 5;
  const int wm    = wave & 3;    // 4 waves along M
  const int wn    = wave >> 2;   // 2 waves along N
  const int half  = lane >> 4;
  const int lcol  = lane & 15;
  const int mBase = blockIdx.y * TILE_M;
  const int nBase = blockIdx.x * TILE_N;

  const v8f vzero = {0.f, 0.f, 0.f, 0.f, 0.f, 0.f, 0.f, 0.f};
  v8f acc[2][4];
#pragma unroll
  for (int mi = 0; mi < 2; ++mi)
#pragma unroll
    for (int ni = 0; ni < 4; ++ni) acc[mi][ni] = vzero;

  for (int k0 = 0; k0 < K; k0 += TILE_K) {
    __syncthreads();
    // ---- stage A tile (f32 -> bf16), [m][k]
#pragma unroll
    for (int i = 0; i < 4; ++i) {
      int idx = tid + 256 * i;           // 1024 float4 covers 128x32
      int r   = idx >> 3;
      int kc  = (idx & 7) << 2;
      float4 v = *(const float4*)(A + (size_t)(mBase + r) * K + (k0 + kc));
      v4bf h; h[0] = (__bf16)v.x; h[1] = (__bf16)v.y; h[2] = (__bf16)v.z; h[3] = (__bf16)v.w;
      *(v4bf*)&As[r][kc] = h;
    }
    // ---- stage B tile as [n][k]
    if constexpr (B_NMAJOR) {
#pragma unroll
      for (int i = 0; i < 4; ++i) {
        int idx = tid + 256 * i;
        int r   = idx >> 3;
        int kc  = (idx & 7) << 2;
        float4 v = *(const float4*)(Bsrc + (size_t)(nBase + r) * K + (k0 + kc));
        v4bf h; h[0] = (__bf16)v.x; h[1] = (__bf16)v.y; h[2] = (__bf16)v.z; h[3] = (__bf16)v.w;
        *(v4bf*)&Bs[r][kc] = h;
      }
    } else {
#pragma unroll
      for (int i = 0; i < 4; ++i) {
        int idx = tid + 256 * i;           // 1024 float4 covers 32x128
        int kr  = idx >> 5;
        int nc  = (idx & 31) << 2;
        float4 v = *(const float4*)(Bsrc + (size_t)(k0 + kr) * N + (nBase + nc));
        Bs[nc + 0][kr] = (__bf16)v.x;
        Bs[nc + 1][kr] = (__bf16)v.y;
        Bs[nc + 2][kr] = (__bf16)v.z;
        Bs[nc + 3][kr] = (__bf16)v.w;
      }
    }
    __syncthreads();

    // ---- fragments (ISA 16-bit A/B wave32 layouts)
    v16bf afr[2], bfr[4];
    const int akb = half ? 8 : 0;          // lanes 16-31: K 8..15 / 24..31
#pragma unroll
    for (int mi = 0; mi < 2; ++mi) {
      int r = wm * 32 + mi * 16 + lcol;
      v8bf lo = *(const v8bf*)&As[r][akb];
      v8bf hi = *(const v8bf*)&As[r][akb + 16];
      afr[mi] = __builtin_shufflevector(lo, hi, 0,1,2,3,4,5,6,7,8,9,10,11,12,13,14,15);
    }
    const int bkb = half ? 8 : 0;          // lanes 16-31: K 8..15 / 24..31
#pragma unroll
    for (int ni = 0; ni < 4; ++ni) {
      int c = wn * 64 + ni * 16 + lcol;
      v8bf lo = *(const v8bf*)&Bs[c][bkb];
      v8bf hi = *(const v8bf*)&Bs[c][bkb + 16];
      bfr[ni] = __builtin_shufflevector(lo, hi, 0,1,2,3,4,5,6,7,8,9,10,11,12,13,14,15);
    }
#pragma unroll
    for (int mi = 0; mi < 2; ++mi)
#pragma unroll
      for (int ni = 0; ni < 4; ++ni)
        acc[mi][ni] = __builtin_amdgcn_wmma_f32_16x16x32_bf16(
            false, afr[mi], false, bfr[ni], (short)0, acc[mi][ni], false, false);
  }

  // ---- epilogue
  if constexpr (EPI == EPI_ARGMIN) {
    // dist-partial = ||e_n||^2 - 2*dot (row constant ||z||^2 dropped: argmin-invariant)
    float een[4];
#pragma unroll
    for (int ni = 0; ni < 4; ++ni)
      een[ni] = rowsq[nBase + wn * 64 + ni * 16 + lcol];
#pragma unroll
    for (int mi = 0; mi < 2; ++mi) {
#pragma unroll
      for (int j = 0; j < 8; ++j) {
        unsigned long long best = ~0ull;
#pragma unroll
        for (int ni = 0; ni < 4; ++ni) {
          int n = nBase + wn * 64 + ni * 16 + lcol;
          float val = een[ni] - 2.0f * acc[mi][ni][j];
          unsigned long long key = ((unsigned long long)float_key(val) << 32) | (unsigned)n;
          best = key < best ? key : best;
        }
#pragma unroll
        for (int off = 1; off < 16; off <<= 1) {   // reduce across the 16 cols/row
          unsigned long long o = shfl_xor_u64(best, off);
          best = o < best ? o : best;
        }
        if (lcol == 0) {
          int m = mBase + wm * 32 + mi * 16 + half * 8 + j;
          atomicMin(&amin[m], best);
        }
      }
    }
  } else {
#pragma unroll
    for (int mi = 0; mi < 2; ++mi) {
#pragma unroll
      for (int ni = 0; ni < 4; ++ni) {
        int n = nBase + wn * 64 + ni * 16 + lcol;
        float bn = bias[n];
#pragma unroll
        for (int j = 0; j < 8; ++j) {
          int m = mBase + wm * 32 + mi * 16 + half * 8 + j;
          float v = acc[mi][ni][j] + bn;
          if constexpr (EPI == EPI_RELU_BIAS) {
            v = v > 0.0f ? v : 0.0f;
          } else if constexpr (EPI == EPI_RES) {
            v = (v > 0.0f ? v : 0.0f) + Cadd[(size_t)m * N + n];
          }
          Out[(size_t)m * N + n] = v;
        }
      }
    }
  }
}

__global__ __launch_bounds__(256)
void row_sqnorm_kernel(const float* __restrict__ E, float* __restrict__ rowsq) {
  int row  = blockIdx.x * 8 + (threadIdx.x >> 5);
  int lane = threadIdx.x & 31;
  const float* p = E + (size_t)row * 256;
  float s = 0.0f;
#pragma unroll
  for (int i = 0; i < 8; ++i) { float v = p[lane + 32 * i]; s += v * v; }
#pragma unroll
  for (int off = 16; off > 0; off >>= 1) s += __shfl_xor(s, off, 32);
  if (lane == 0) rowsq[row] = s;
}

__global__ void init_u64_kernel(unsigned long long* p, int n) {
  int i = blockIdx.x * blockDim.x + threadIdx.x;
  if (i < n) p[i] = ~0ull;
}

// z_q = E[argmin]; vq_loss = (1+beta)*||q - z||^2 (exact f32)
__global__ __launch_bounds__(256)
void vq_gather_kernel(const float* __restrict__ z,
                      const unsigned long long* __restrict__ amin,
                      const float* __restrict__ E,
                      float* __restrict__ zq_out,
                      float* __restrict__ loss_out) {
  int m = blockIdx.x;
  int t = threadIdx.x;                       // 256 == D
  unsigned idx = (unsigned)(amin[m] & 0xFFFFFFFFull);
  float q  = E[(size_t)idx * 256 + t];
  float zv = z[(size_t)m * 256 + t];
  zq_out[(size_t)m * 256 + t] = q;
  float d = q - zv;
  float s = d * d;
#pragma unroll
  for (int off = 16; off > 0; off >>= 1) s += __shfl_xor(s, off, 32);
  __shared__ float part[8];
  if ((t & 31) == 0) part[t >> 5] = s;
  __syncthreads();
  if (t == 0) {
    float tot = 0.0f;
#pragma unroll
    for (int w = 0; w < 8; ++w) tot += part[w];
    loss_out[m] = 1.25f * tot;
  }
}

extern "C" void kernel_launch(void* const* d_in, const int* in_sizes, int n_in,
                              void* d_out, int out_size, void* d_ws, size_t ws_size,
                              hipStream_t stream) {
  (void)in_sizes; (void)n_in; (void)out_size; (void)ws_size;
  const float* x_a      = (const float*)d_in[0];
  const float* x_b      = (const float*)d_in[1];
  const float* enc_a_w  = (const float*)d_in[2];
  const float* enc_a_b  = (const float*)d_in[3];
  const float* enc_a_rw = (const float*)d_in[4];
  const float* enc_a_rb = (const float*)d_in[5];
  const float* enc_b_w  = (const float*)d_in[6];
  const float* enc_b_b  = (const float*)d_in[7];
  const float* enc_b_rw = (const float*)d_in[8];
  const float* enc_b_rb = (const float*)d_in[9];
  const float* dec_a_rw = (const float*)d_in[10];
  const float* dec_a_rb = (const float*)d_in[11];
  const float* dec_a_w  = (const float*)d_in[12];
  const float* dec_a_b  = (const float*)d_in[13];
  const float* dec_b_rw = (const float*)d_in[14];
  const float* dec_b_rb = (const float*)d_in[15];
  const float* dec_b_w  = (const float*)d_in[16];
  const float* dec_b_b  = (const float*)d_in[17];
  const float* codebook = (const float*)d_in[18];

  const int B = 8192, D = 256, KCB = 8192, DA = 4096, DB = 2048;

  float* out = (float*)d_out;
  float* z_a = out;
  float* z_b = z_a + (size_t)B * D;
  float* la  = z_b + (size_t)B * D;
  float* lb  = la + B;
  float* ra  = lb + B;
  float* rb  = ra + (size_t)B * DA;
  float* cra = rb + (size_t)B * DB;
  float* crb = cra + (size_t)B * DA;

  unsigned long long* amin_a = (unsigned long long*)d_ws;
  unsigned long long* amin_b = amin_a + B;
  float* rowsq = (float*)(amin_b + B);
  float* buf0  = rowsq + KCB;
  float* buf1  = buf0 + (size_t)B * D;

  row_sqnorm_kernel<<<KCB / 8, 256, 0, stream>>>(codebook, rowsq);
  init_u64_kernel<<<(2 * B + 255) / 256, 256, 0, stream>>>(amin_a, 2 * B);

  dim3 blk(256);
  auto g = [](int M, int N) { return dim3((unsigned)(N / TILE_N), (unsigned)(M / TILE_M)); };

  // ---- encoder A -> VQ A (frees buffers before encoder B)
  gemm_bf16_wmma<EPI_RELU_BIAS, false><<<g(B, D), blk, 0, stream>>>(
      x_a, enc_a_w, enc_a_b, nullptr, buf0, nullptr, nullptr, B, D, DA);
  gemm_bf16_wmma<EPI_RES, false><<<g(B, D), blk, 0, stream>>>(
      buf0, enc_a_rw, enc_a_rb, buf0, buf1, nullptr, nullptr, B, D, D);
  gemm_bf16_wmma<EPI_RES, false><<<g(B, D), blk, 0, stream>>>(
      buf1, enc_a_rw + D * D, enc_a_rb + D, buf1, buf0, nullptr, nullptr, B, D, D);
  gemm_bf16_wmma<EPI_ARGMIN, true><<<g(B, KCB), blk, 0, stream>>>(
      buf0, codebook, nullptr, nullptr, nullptr, rowsq, amin_a, B, KCB, D);
  vq_gather_kernel<<<B, 256, 0, stream>>>(buf0, amin_a, codebook, z_a, la);

  // ---- encoder B -> VQ B
  gemm_bf16_wmma<EPI_RELU_BIAS, false><<<g(B, D), blk, 0, stream>>>(
      x_b, enc_b_w, enc_b_b, nullptr, buf0, nullptr, nullptr, B, D, DB);
  gemm_bf16_wmma<EPI_RES, false><<<g(B, D), blk, 0, stream>>>(
      buf0, enc_b_rw, enc_b_rb, buf0, buf1, nullptr, nullptr, B, D, D);
  gemm_bf16_wmma<EPI_RES, false><<<g(B, D), blk, 0, stream>>>(
      buf1, enc_b_rw + D * D, enc_b_rb + D, buf1, buf0, nullptr, nullptr, B, D, D);
  gemm_bf16_wmma<EPI_ARGMIN, true><<<g(B, KCB), blk, 0, stream>>>(
      buf0, codebook, nullptr, nullptr, nullptr, rowsq, amin_b, B, KCB, D);
  vq_gather_kernel<<<B, 256, 0, stream>>>(buf0, amin_b, codebook, z_b, lb);

  // ---- decoders: dec_a(z_a), dec_b(z_b), dec_a(z_b), dec_b(z_a)
  const float* zin[4]  = {z_a, z_b, z_b, z_a};
  const float* drw[4]  = {dec_a_rw, dec_b_rw, dec_a_rw, dec_b_rw};
  const float* drb[4]  = {dec_a_rb, dec_b_rb, dec_a_rb, dec_b_rb};
  const float* dw[4]   = {dec_a_w, dec_b_w, dec_a_w, dec_b_w};
  const float* db[4]   = {dec_a_b, dec_b_b, dec_a_b, dec_b_b};
  float*       dout[4] = {ra, rb, cra, crb};
  const int    dn[4]   = {DA, DB, DA, DB};
  for (int i = 0; i < 4; ++i) {
    gemm_bf16_wmma<EPI_RES, false><<<g(B, D), blk, 0, stream>>>(
        zin[i], drw[i], drb[i], zin[i], buf0, nullptr, nullptr, B, D, D);
    gemm_bf16_wmma<EPI_RES, false><<<g(B, D), blk, 0, stream>>>(
        buf0, drw[i] + D * D, drb[i] + D, buf0, buf1, nullptr, nullptr, B, D, D);
    gemm_bf16_wmma<EPI_BIAS, false><<<g(B, dn[i]), blk, 0, stream>>>(
        buf1, dw[i], db[i], nullptr, dout[i], nullptr, nullptr, B, dn[i], D);
  }
}
